// SparseAttention_26465588478184
// MI455X (gfx1250) — matrix operand order repaired and checked by reading, not compiled, not run on hardware
//
#include <hip/hip_runtime.h>
#include <math.h>

// MI455X / gfx1250, wave32. GEMM stages via v_wmma_f32_16x16x32_f16 with
// fragment-native operand layouts (each lane's 16 halfs contiguous -> b128 loads).

typedef __attribute__((ext_vector_type(16))) _Float16 v16h;
typedef __attribute__((ext_vector_type(8)))  float    v8f;

static constexpr int B_ = 4, C_ = 256, N_ = 2048, W_ = 8;
static constexpr int BL = 128;   // tokens per block
static constexpr int NB = 15;    // blocks kept
static constexpr int G_ = 120;   // NB*W heads / perm size
static constexpr int GP = 128;   // G padded to multiple of 32
#define TEMP_ 0.7f
#define EPS_  1e-6f

// Fragment-layout inverses (16-bit WMMA operands, wave32):
//  A (16x32 MxK): lane = (m&15) | ((k>>3 & 1)<<4); elem = ((k>>4)<<3) | (k&7)
//  B (32x16 KxN): lane = (n&15) | ((k>>3 & 1)<<4); elem = ((k>>4)<<3) | (k&7)
//  C/D (16x16):   VGPR r -> row M = r + 8*(lane>>4), col N = lane%16
__device__ __forceinline__ int aLane(int m, int k) { return (m & 15) | (((k >> 3) & 1) << 4); }
__device__ __forceinline__ int aElem(int k)        { return (((k >> 4) & 1) << 3) | (k & 7); }
__device__ __forceinline__ int bLane(int n, int k) { return (n & 15) | (((k >> 3) & 1) << 4); }
__device__ __forceinline__ int bElem(int k)        { return (((k >> 4) & 1) << 3) | (k & 7); }

__device__ __forceinline__ v16h ldFrag(const _Float16* p) { return *(const v16h*)p; }

__device__ __forceinline__ v8f wmma16(v16h a, v16h b, v8f c) {
  return __builtin_amdgcn_wmma_f32_16x16x32_f16(false, a, false, b, (short)0, c, false, false);
}

union Pack8 { _Float16 h[8]; uint4 v4; };

// ---------------- K1: channel means of blockified q,k -> sq,sk [B][BL][G] ----------------
__global__ void k_mean(const float* __restrict__ q, const float* __restrict__ k,
                       float* __restrict__ sq, float* __restrict__ sk) {
  int t = blockIdx.x * blockDim.x + threadIdx.x;
  int total = B_ * BL * G_;
  if (t >= 2 * total) return;
  const float* src = (t < total) ? q : k;
  float* dst = (t < total) ? sq : sk;
  int e = (t < total) ? t : t - total;
  int g = e % G_, l = (e / G_) % BL, b = e / (G_ * BL);
  int n = (g >> 3) * BL + l, w = g & 7;
  const float* p = src + ((b * C_) * N_ + n) * W_ + w;
  float s = 0.f;
  for (int c = 0; c < C_; ++c) s += p[c * N_ * W_];
  dst[e] = s * (1.0f / C_);
}

// ---------------- K2: R = sq^T sk * C^-1/2; r0 = (log(relu(R)+eps)+gumbel)/T ----------------
__global__ void k_logits(const float* __restrict__ sq, const float* __restrict__ sk,
                         const float* __restrict__ gum, float* __restrict__ rmat) {
  int i = blockIdx.x, b = blockIdx.y, j = threadIdx.x;
  if (j >= G_) return;
  const float* pq = sq + b * BL * G_ + i;
  const float* pk = sk + b * BL * G_ + j;
  float acc = 0.f;
  for (int l = 0; l < BL; ++l) acc += pq[l * G_] * pk[l * G_];
  float R = acc * 0.0625f;  // 256^-0.5
  rmat[(b * G_ + i) * G_ + j] =
      (logf(fmaxf(R, 0.f) + EPS_) + gum[(b * G_ + i) * G_ + j]) * (1.0f / TEMP_);
}

// ---------------- K3: Sinkhorn -> perm written in BOTH fragment layouts (zero-padded) --------
// permB: [b][kt(4)][nt(8)][lane(32)][16]  (B operand of sort GEMM, K=g', N=f)
// permA: [b][mt(8)][kt(4)][lane(32)][16]  (A operand of unsort GEMM, M=i, K=g)
__global__ void k_sinkhorn(const float* __restrict__ rmat,
                           _Float16* __restrict__ permB, _Float16* __restrict__ permA) {
  __shared__ float m[G_ * (G_ + 1)];
  int b = blockIdx.x, t = threadIdx.x;  // 128 threads
  for (int idx = t; idx < G_ * G_; idx += 128) {
    int i = idx / G_, j = idx % G_;
    m[i * (G_ + 1) + j] = rmat[b * G_ * G_ + idx];
  }
  __syncthreads();
  for (int it = 0; it < 8; ++it) {
    if (t < G_) {  // row LSE
      float mx = -1e30f;
      for (int j = 0; j < G_; ++j) mx = fmaxf(mx, m[t * (G_ + 1) + j]);
      float s = 0.f;
      for (int j = 0; j < G_; ++j) s += expf(m[t * (G_ + 1) + j] - mx);
      float lse = logf(s) + mx;
      for (int j = 0; j < G_; ++j) m[t * (G_ + 1) + j] -= lse;
    }
    __syncthreads();
    if (t < G_) {  // col LSE
      float mx = -1e30f;
      for (int i = 0; i < G_; ++i) mx = fmaxf(mx, m[i * (G_ + 1) + t]);
      float s = 0.f;
      for (int i = 0; i < G_; ++i) s += expf(m[i * (G_ + 1) + t] - mx);
      float lse = logf(s) + mx;
      for (int i = 0; i < G_; ++i) m[i * (G_ + 1) + t] -= lse;
    }
    __syncthreads();
  }
  for (int idx = t; idx < GP * GP; idx += 128) {
    int kk = idx >> 7, nn = idx & 127;  // (row, col) of perm
    float pv = (kk < G_ && nn < G_) ? expf(m[kk * (G_ + 1) + nn]) : 0.f;
    _Float16 hv = (_Float16)pv;
    {  // B layout: K=kk, N=nn
      int kt = kk >> 5, kl = kk & 31;
      permB[b * GP * GP + ((kt * 8 + (nn >> 4)) * 32 + bLane(nn, kl)) * 16 + bElem(kl)] = hv;
    }
    {  // A layout: M=kk, K=nn
      int kt = nn >> 5, kl = nn & 31;
      permA[b * GP * GP + (((kk >> 4) * 4 + kt) * 32 + aLane(kk, kl)) * 16 + aElem(kl)] = hv;
    }
  }
}

// ---------------- K3b: pack blockified q into A-fragment layout per head ----------------
// qA: [b][g][it(8)][kt(8)][lane(32)][16]   (M=i token-in-block, K=c channel)
// An aligned 8-channel group maps to 8 contiguous fragment halfs -> one b128 store.
__global__ void k_qpack(const float* __restrict__ q, _Float16* __restrict__ qA) {
  int g = blockIdx.x, b = blockIdx.y, t = threadIdx.x;  // 256 threads
  int n0 = (g >> 3) * BL, wq = g & 7;
  _Float16* dst = qA + (size_t)(b * G_ + g) * 64 * 512;
  for (int u = t; u < BL * 32; u += 256) {  // unit = (i, 8-channel group)
    int i = u & 127, cg = u >> 7;
    int c0 = cg * 8;
    Pack8 pk;
#pragma unroll
    for (int cc = 0; cc < 8; ++cc)
      pk.h[cc] = (_Float16)q[((b * C_ + c0 + cc) * N_ + n0 + i) * W_ + wq];
    int kt = c0 >> 5, kl0 = c0 & 31;
    int e0 = ((kl0 >> 4) & 1) << 3;
    *(uint4*)&dst[(size_t)(((i >> 4) * 8 + kt) * 32 + aLane(i, kl0)) * 16 + e0] = pk.v4;
  }
}

// ---------------- K4: sort GEMM, templated on which tensor (k vs v) ----------------
// WHICH==0: ksortB [b][f][kt(8,K=c)][nt(8,N=l)][lane][16]
// WHICH==1: vsortB [b][f][kt(4,K=l)][nt(16,N=c)][lane][16]
template <int WHICH>
__global__ void k_sort_t(const float* __restrict__ src, const _Float16* __restrict__ permB,
                         _Float16* __restrict__ outB) {
  __shared__ __align__(32) _Float16 permLds[GP * GP];   // 32 KB, fragment layout
  __shared__ __align__(32) _Float16 aFrag[4 * 32 * 16]; // 4 KB: A frags of this 16-row stripe
  int b = blockIdx.y;
  int m0 = blockIdx.x * 16;
  int t = threadIdx.x;  // 128 threads, 4 waves
  {  // vectorized LDS copy of swizzled perm
    const uint4* s = (const uint4*)(permB + b * GP * GP);
    uint4* d = (uint4*)permLds;
    for (int idx = t; idx < GP * GP / 8; idx += 128) d[idx] = s[idx];
  }
  // A staging: blockify is contiguous over aligned 8-groups of g' (W dim) -> float4 x2 + b128 ds store
  for (int u = t; u < 16 * 16; u += 128) {  // (row r, 8-group gg)
    int r = u & 15, g0 = (u >> 4) * 8;
    int mm = m0 + r, c = mm >> 7, l = mm & 127;
    Pack8 pk;
    if (g0 < G_) {
      const float* sp = &src[((b * C_ + c) * N_ + (g0 >> 3) * BL + l) * W_];
      float4 lo = *(const float4*)sp;
      float4 hi = *(const float4*)(sp + 4);
      pk.h[0] = (_Float16)lo.x; pk.h[1] = (_Float16)lo.y;
      pk.h[2] = (_Float16)lo.z; pk.h[3] = (_Float16)lo.w;
      pk.h[4] = (_Float16)hi.x; pk.h[5] = (_Float16)hi.y;
      pk.h[6] = (_Float16)hi.z; pk.h[7] = (_Float16)hi.w;
    } else {
      pk.v4 = make_uint4(0, 0, 0, 0);
    }
    int kt = g0 >> 5, kl0 = g0 & 31;
    int e0 = ((kl0 >> 4) & 1) << 3;
    *(uint4*)&aFrag[(kt * 32 + aLane(r, kl0)) * 16 + e0] = pk.v4;
  }
  __syncthreads();
  int wave = t >> 5, lane = t & 31, h = lane >> 4, nlo = lane & 15;
  int c = (m0 + 8 * h) >> 7, l0 = (m0 + 8 * h) & 127;  // constant across the 8 rows per lane
  v8f zero = {};
  for (int nt = wave; nt < 8; nt += 4) {
    v8f acc = zero;
#pragma unroll
    for (int kt = 0; kt < 4; ++kt) {
      v16h a  = ldFrag(&aFrag[(kt * 32 + lane) * 16]);
      v16h bf = ldFrag(&permLds[((kt * 8 + nt) * 32 + lane) * 16]);
      acc = wmma16(a, bf, acc);
    }
    int n = nt * 16 + nlo;  // head f
    if (n < G_) {
      if (WHICH == 0) {  // K=c, N=l; rows step bLane by 1 -> stride 16 halfs
        int kto = c >> 5, klc = c & 31;
        _Float16* p = &outB[(size_t)(((b * G_ + n) * 8 + kto) * 8 + (l0 >> 4)) * 512 +
                            bLane(l0, klc) * 16 + bElem(klc)];
#pragma unroll
        for (int r = 0; r < 8; ++r) p[r * 16] = (_Float16)acc[r];
      } else {           // K=l, N=c; rows step bElem by 1 -> 8 contiguous halfs = b128 store
        int kto = l0 >> 5, kl0s = l0 & 31;
        Pack8 pk;
#pragma unroll
        for (int r = 0; r < 8; ++r) pk.h[r] = (_Float16)acc[r];
        *(uint4*)&outB[(size_t)(((b * G_ + n) * 4 + kto) * 16 + (c >> 4)) * 512 +
                       bLane(c, kl0s) * 16 + bElem(kl0s)] = pk.v4;
      }
    }
  }
}

// ---------------- K5: per-head attention (WMMA + in-register softmax) ----------------
// valB: [b][kt(4,K=g)][nt(2048,N=l*256+c)][lane][16]  (B operand of unsort GEMM)
__global__ void k_attn(const _Float16* __restrict__ qA, const _Float16* __restrict__ ksortB,
                       const _Float16* __restrict__ vsortB, _Float16* __restrict__ valB) {
  __shared__ __align__(32) _Float16 attnA[8 * 4 * 32 * 16];  // 32 KB: per-wave A frags
  int g = blockIdx.x, b = blockIdx.y, t = threadIdx.x;  // 256 threads, 8 waves
  int wave = t >> 5, lane = t & 31, h = lane >> 4, nlo = lane & 15;
  size_t hb = (size_t)(b * G_ + g) * 64 * 512;
  const _Float16* qAg = qA + hb;
  const _Float16* kBg = ksortB + hb;
  const _Float16* vBg = vsortB + hb;
  v8f zero = {};
  v8f acc[8];
#pragma unroll
  for (int jt = 0; jt < 8; ++jt) acc[jt] = zero;
  // scores = qt @ kt^T  (K = C = 256); wave owns rows [16*wave, 16*wave+16)
  for (int kt = 0; kt < 8; ++kt) {
    v16h a = ldFrag(&qAg[((wave * 8 + kt) * 32 + lane) * 16]);
    if (kt < 7) {  // prefetch next K-step's operands -> global_prefetch_b8
      __builtin_prefetch(&qAg[((wave * 8 + kt + 1) * 32 + lane) * 16], 0, 0);
      __builtin_prefetch(&kBg[(((kt + 1) * 8) * 32 + lane) * 16], 0, 0);
    }
#pragma unroll
    for (int jt = 0; jt < 8; ++jt) {
      v16h bf = ldFrag(&kBg[((kt * 8 + jt) * 32 + lane) * 16]);
      acc[jt] = wmma16(a, bf, acc[jt]);
    }
  }
  // softmax over j; row i = 16*wave + r + 8h lives in a 16-lane half-group
  const float invT = 1.0f / TEMP_;
  float rmax[8], rsum[8];
#pragma unroll
  for (int r = 0; r < 8; ++r) rmax[r] = -1e30f;
#pragma unroll
  for (int jt = 0; jt < 8; ++jt)
#pragma unroll
    for (int r = 0; r < 8; ++r) rmax[r] = fmaxf(rmax[r], acc[jt][r] * invT);
#pragma unroll
  for (int r = 0; r < 8; ++r) {
    for (int off = 1; off < 16; off <<= 1)
      rmax[r] = fmaxf(rmax[r], __shfl_xor(rmax[r], off, 32));
    rsum[r] = 0.f;
  }
#pragma unroll
  for (int jt = 0; jt < 8; ++jt)
#pragma unroll
    for (int r = 0; r < 8; ++r) {
      float p = expf(acc[jt][r] * invT - rmax[r]);
      acc[jt][r] = p;
      rsum[r] += p;
    }
#pragma unroll
  for (int r = 0; r < 8; ++r)
    for (int off = 1; off < 16; off <<= 1)
      rsum[r] += __shfl_xor(rsum[r], off, 32);
  // write attn into this wave's A-fragment slots (swizzled scatter, 2B stores)
#pragma unroll
  for (int jt = 0; jt < 8; ++jt)
#pragma unroll
    for (int r = 0; r < 8; ++r) {
      int m = r + 8 * h, j = jt * 16 + nlo;
      int kt = j >> 5, kl = j & 31;
      attnA[((wave * 4 + kt) * 32 + aLane(m, kl)) * 16 + aElem(kl)] =
          (_Float16)(acc[jt][r] / rsum[r]);
    }
  __syncthreads();
  // val = attn @ vt (K = 128, N = C = 256); scatter into unsort B-fragment layout
  int ktv = g >> 5, klv = g & 31;
  int laneV = bLane(nlo, klv), eV = bElem(klv);
  _Float16* vb = valB + (size_t)(b * 4 + ktv) * 2048 * 512;
  for (int nt = 0; nt < 16; ++nt) {
    v8f acc2 = zero;
#pragma unroll
    for (int kt = 0; kt < 4; ++kt) {
      v16h a  = ldFrag(&attnA[((wave * 4 + kt) * 32 + lane) * 16]);
      v16h bf = ldFrag(&vBg[((kt * 16 + nt) * 32 + lane) * 16]);
      acc2 = wmma16(a, bf, acc2);
    }
    // rows i = 16*wave + 8h + r -> constant stride 8192 halfs (foldable offsets)
    _Float16* p = vb + (size_t)(((wave * 16 + 8 * h) * 16 + nt) * 512) + laneV * 16 + eV;
#pragma unroll
    for (int r = 0; r < 8; ++r) p[r * 8192] = (_Float16)acc2[r];
  }
}

// ---------------- K6: opv = perm @ val (WMMA), scatter to output layout ----------------
__global__ void k_unsort(const _Float16* __restrict__ permA, const _Float16* __restrict__ valB,
                         float* __restrict__ out) {
  int b = blockIdx.y, ntile = blockIdx.x, t = threadIdx.x;  // 128 threads, 4 waves
  int wave = t >> 5, lane = t & 31, h = lane >> 4, nlo = lane & 15;
  v8f zero = {};
  v8f acc[2] = {zero, zero};
#pragma unroll
  for (int kt = 0; kt < 4; ++kt) {
    v16h bf = ldFrag(&valB[(size_t)((b * 4 + kt) * 2048 + ntile) * 512 + lane * 16]);
#pragma unroll
    for (int mi = 0; mi < 2; ++mi) {
      int mt = wave + 4 * mi;
      v16h a = ldFrag(&permA[(size_t)(b * 32 + mt * 4 + kt) * 512 + lane * 16]);
      acc[mi] = wmma16(a, bf, acc[mi]);
    }
  }
  int n = ntile * 16 + nlo, l = n >> 8, c = n & 255;
#pragma unroll
  for (int mi = 0; mi < 2; ++mi) {
    int mt = wave + 4 * mi;
    int ibase = mt * 16 + 8 * h;
    if (ibase < G_) {  // uniform per (mt,h): rows ibase..ibase+7 all valid or all padding
#pragma unroll
      for (int r = 0; r < 8; ++r) {
        int i = ibase + r;
        int blk = i >> 3, wq = i & 7;
        out[((b * W_ + wq) * N_ + blk * BL + l) * C_ + c] = acc[mi][r];
      }
    }
  }
}

// ---------------- K7: last-block passthrough out[b,w,1920+l,c] = v[b,c,1920+l,w] ------------
__global__ void k_tail(const float* __restrict__ v, float* __restrict__ out) {
  int idx = blockIdx.x * blockDim.x + threadIdx.x;
  if (idx >= B_ * W_ * BL * C_) return;
  int c = idx & 255, l = (idx >> 8) & 127, wq = (idx >> 15) & 7, b = idx >> 18;
  out[((b * W_ + wq) * N_ + NB * BL + l) * C_ + c] =
      v[((b * C_ + c) * N_ + NB * BL + l) * W_ + wq];
}

extern "C" void kernel_launch(void* const* d_in, const int* in_sizes, int n_in,
                              void* d_out, int out_size, void* d_ws, size_t ws_size,
                              hipStream_t stream) {
  (void)in_sizes; (void)n_in; (void)out_size; (void)ws_size;
  const float* q   = (const float*)d_in[0];
  const float* k   = (const float*)d_in[1];
  const float* v   = (const float*)d_in[2];
  const float* gum = (const float*)d_in[3];
  float* out = (float*)d_out;

  // Workspace carve (~129 MB total)
  char* ws = (char*)d_ws;
  size_t off = 0;
  auto carve = [&](size_t bytes) -> void* {
    void* p = (void*)(ws + off);
    off += (bytes + 255) & ~(size_t)255;
    return p;
  };
  float*    sq    = (float*)carve((size_t)B_ * BL * G_ * sizeof(float));
  float*    sk    = (float*)carve((size_t)B_ * BL * G_ * sizeof(float));
  float*    rmat  = (float*)carve((size_t)B_ * G_ * G_ * sizeof(float));
  _Float16* permB = (_Float16*)carve((size_t)B_ * GP * GP * 2);
  _Float16* permA = (_Float16*)carve((size_t)B_ * GP * GP * 2);
  _Float16* qA    = (_Float16*)carve((size_t)B_ * G_ * 64 * 512 * 2);
  _Float16* ksortB= (_Float16*)carve((size_t)B_ * G_ * 64 * 512 * 2);
  _Float16* vsortB= (_Float16*)carve((size_t)B_ * G_ * 64 * 512 * 2);
  size_t valBytes = (size_t)B_ * 4 * 2048 * 512 * 2;
  _Float16* valB  = (_Float16*)carve(valBytes);

  k_mean<<<(2 * B_ * BL * G_ + 255) / 256, 256, 0, stream>>>(q, k, sq, sk);
  k_logits<<<dim3(G_, B_), 128, 0, stream>>>(sq, sk, gum, rmat);
  k_sinkhorn<<<B_, 128, 0, stream>>>(rmat, permB, permA);
  k_qpack<<<dim3(G_, B_), 256, 0, stream>>>(q, qA);
  k_sort_t<0><<<dim3(C_ * BL / 16, B_), 128, 0, stream>>>(k, permB, ksortB);
  k_sort_t<1><<<dim3(C_ * BL / 16, B_), 128, 0, stream>>>(v, permB, vsortB);
  hipMemsetAsync(valB, 0, valBytes, stream);  // zero K-padding rows (g=120..127)
  k_attn<<<dim3(G_, B_), 256, 0, stream>>>(qA, ksortB, vsortB, valB);
  k_unsort<<<dim3(BL * C_ / 16, B_), 128, 0, stream>>>(permA, valB, out);
  k_tail<<<(B_ * W_ * BL * C_ + 255) / 256, 256, 0, stream>>>(v, out);
}
